// MultiHeadAttention_24240795419423
// MI455X (gfx1250) — matrix operand order repaired and checked by reading, not compiled, not run on hardware
//
#include <hip/hip_runtime.h>
#include <hip/hip_bf16.h>

#define B_SZ   2
#define T_LEN  2048
#define S_LEN  2048
#define EMB    1024
#define NH     16
#define DH     64
#define SCALE_F 0.125f
#define CLAMP_V 50000.0f

typedef __attribute__((ext_vector_type(16))) __bf16 bf16x16;
typedef __attribute__((ext_vector_type(8)))  float  f32x8;

struct alignas(16) U128 { unsigned x, y, z, w; };

union FragB {
  bf16x16 v;
  __bf16  h[16];
  U128    q[2];
};

__device__ inline __bf16 f2bf(float x) {
  union { float f; unsigned u; } a; a.f = x;
  unsigned r = a.u + 0x7FFFu + ((a.u >> 16) & 1u);   // round-to-nearest-even
  union { unsigned short us; __bf16 b; } o;
  o.us = (unsigned short)(r >> 16);
  return o.b;
}

__device__ inline f32x8 wmma_bf16(bf16x16 a, bf16x16 b, f32x8 c) {
  // D = A(16x32 bf16) * B(32x16 bf16) + C(16x16 f32)
  return __builtin_amdgcn_wmma_f32_16x16x32_bf16(
      /*neg_a=*/false, a, /*neg_b=*/false, b,
      /*c_mod=*/(short)0, c, /*reuse_a=*/false, /*reuse_b=*/false);
}

// ---- fragment loaders (wave32 layouts per CDNA5 ISA 7.12.2) -----------------
// A 16x32 bf16, row-major source, leading dim lda. lane = row (lane&15);
// lo lanes hold K in [0..7]+[16..23], hi lanes K in [8..15]+[24..31].
__device__ inline bf16x16 load_a_bf16(const __bf16* A, int lda, int row, int k0, int hi) {
  FragB f;
  const __bf16* p = A + (size_t)row * lda + k0 + hi * 8;
  f.q[0] = *(const U128*)p;          // elems 0..7  -> K = k0+hi*8    .. +7
  f.q[1] = *(const U128*)(p + 16);   // elems 8..15 -> K = k0+16+hi*8 .. +7
  return f.v;
}

// B 32x16 for NT gemm: source row-major [N,K]; wmmaB(k,n) = Bm[n0+n][k0+k].
// lane column = lane&15; lo lanes K=[k0..k0+7]+[k0+16..k0+23], hi lanes K=[k0+8..k0+15]+[k0+24..k0+31].
__device__ inline bf16x16 load_b_nt(const __bf16* Bm, int ldb, int n0, int k0, int lr, int hi) {
  FragB f;
  const __bf16* p = Bm + (size_t)(n0 + lr) * ldb + k0 + hi * 8;
  f.q[0] = *(const U128*)p;
  f.q[1] = *(const U128*)(p + 16);
  return f.v;
}

// B 32x16 fragment from an LDS-staged panel laid out [col][k] (32 bf16 per col).
__device__ inline bf16x16 load_b_lds(const __bf16* sW, int col, int hi) {
  FragB f;
  const __bf16* p = sW + col * 32 + hi * 8;
  f.q[0] = *(const U128*)p;
  f.q[1] = *(const U128*)(p + 16);
  return f.v;
}

// A 16x32 from an fp32 row-major source, converting to bf16 in-register.
__device__ inline bf16x16 load_a_f32(const float* A, int lda, int row, int k0, int hi) {
  FragB f;
  const float* p = A + (size_t)row * lda + k0 + hi * 8;
#pragma unroll
  for (int e = 0; e < 8; ++e) {
    f.h[e]     = f2bf(p[e]);
    f.h[8 + e] = f2bf(p[e + 16]);
  }
  return f.v;
}

// ---- kernel 1: fp32 -> bf16 -------------------------------------------------
__global__ void cvt_f32_bf16(const float* __restrict__ src, __bf16* __restrict__ dst, int n) {
  int i = blockIdx.x * blockDim.x + threadIdx.x;
  if (i < n) dst[i] = f2bf(src[i]);
}

// ---- kernel 2: C = (A @ W^T + bias) * alpha --------------------------------
// A: [M,K] bf16 row-major, W: [N,K] bf16 row-major.
// W panel (128 cols x 32 K = 8KB) is staged in LDS per K-step with
// global_load_async_to_lds_b128 (ASYNCcnt path), 4x wave reuse per block.
// mode 0: fp32 row-major [M,N] to outF.
// mode 1: bf16 head-split [B,NH,T,DH] to outH (M = B*T, N = NH*DH).
// mode 2: bf16 head-split transposed [B,NH,DH,S] to outH (for V).
__global__ __launch_bounds__(256)
void gemm_bf16_nt(const __bf16* __restrict__ A, const __bf16* __restrict__ W,
                  const float* __restrict__ bias, float alpha,
                  int M, int N, int K,
                  float* __restrict__ outF, __bf16* __restrict__ outH, int mode) {
  __shared__ __align__(16) __bf16 sW[128 * 32];   // [col][k], 64B per column

  const int tilesN = N >> 7;
  const int tm = blockIdx.x / tilesN;
  const int tn = blockIdx.x % tilesN;
  const int wave = threadIdx.x >> 5, lane = threadIdx.x & 31;
  const int wm = wave >> 1, wn = wave & 1;        // 4x2 waves -> 128x128 block tile
  const int lr = lane & 15, hi = lane >> 4;
  const int row0 = tm * 128 + wm * 32;
  const int col0 = tn * 128 + wn * 64;

  // async-copy assignment: thread copies 32B of the W panel
  const int ccol  = threadIdx.x & 127;            // panel column 0..127
  const int khalf = threadIdx.x >> 7;             // 0/1 -> k 0..15 / 16..31
  const unsigned lds_base = (unsigned)(size_t)&sW[0];
  const unsigned lds_dst  = lds_base + (unsigned)(ccol * 64 + khalf * 32);
  const __bf16* wsrc_base = W + (size_t)(tn * 128 + ccol) * K + khalf * 16;

  f32x8 acc[2][4] = {};
  for (int k0 = 0; k0 < K; k0 += 32) {
    // stage W panel for this K-step into LDS (async, tracked by ASYNCcnt)
    {
      unsigned long long ga = (unsigned long long)(size_t)(wsrc_base + k0);
      asm volatile("global_load_async_to_lds_b128 %0, %1, off"
                   :: "v"(lds_dst), "v"(ga) : "memory");
      asm volatile("global_load_async_to_lds_b128 %0, %1, off offset:16"
                   :: "v"(lds_dst), "v"(ga) : "memory");
      asm volatile("s_wait_asynccnt 0x0" ::: "memory");
    }
    __syncthreads();

    if (k0 + 32 < K)
      __builtin_prefetch(A + (size_t)(row0 + lr) * K + k0 + 32, 0, 3);

    bf16x16 a0 = load_a_bf16(A, K, row0 + lr, k0, hi);
    bf16x16 a1 = load_a_bf16(A, K, row0 + 16 + lr, k0, hi);
#pragma unroll
    for (int j = 0; j < 4; ++j) {
      bf16x16 bj = load_b_lds(sW, wn * 64 + j * 16 + lr, hi);
      acc[0][j] = wmma_bf16(a0, bj, acc[0][j]);
      acc[1][j] = wmma_bf16(a1, bj, acc[1][j]);
    }
    __syncthreads();                              // protect LDS reuse next step
  }

#pragma unroll
  for (int i = 0; i < 2; ++i)
#pragma unroll
    for (int j = 0; j < 4; ++j)
#pragma unroll
      for (int r = 0; r < 8; ++r) {
        int m = row0 + i * 16 + hi * 8 + r;
        int n = col0 + j * 16 + lr;
        float v = acc[i][j][r];
        if (bias) v += bias[n];
        v *= alpha;
        if (mode == 0) {
          outF[(size_t)m * N + n] = v;
        } else {
          int bb = m >> 11, t = m & (T_LEN - 1);
          int h  = n >> 6,  d = n & (DH - 1);
          if (mode == 1)
            outH[(((size_t)(bb * NH + h)) * T_LEN + t) * DH + d] = f2bf(v);
          else // mode 2: transposed per head, [B,NH,DH,S]
            outH[(((size_t)(bb * NH + h)) * DH + d) * S_LEN + t] = f2bf(v);
        }
      }
}

// ---- kernel 3: attn logits = clamp(Qh @ Kh^T) with mask --------------------
__global__ __launch_bounds__(256)
void attn_scores(const __bf16* __restrict__ Qh, const __bf16* __restrict__ Kh,
                 const int* __restrict__ mask, float* __restrict__ attn) {
  const int bh = blockIdx.y;
  const int b  = bh >> 4;
  const int tilesN = S_LEN >> 7;                  // 16
  const int tm = blockIdx.x / tilesN, tn = blockIdx.x % tilesN;
  const int wave = threadIdx.x >> 5, lane = threadIdx.x & 31;
  const int wm = wave >> 1, wn = wave & 1;
  const int lr = lane & 15, hi = lane >> 4;
  const int row0 = tm * 128 + wm * 32;
  const int col0 = tn * 128 + wn * 64;

  const __bf16* Q  = Qh + (size_t)bh * T_LEN * DH;
  const __bf16* Kp = Kh + (size_t)bh * S_LEN * DH;

  f32x8 acc[2][4] = {};
#pragma unroll
  for (int k0 = 0; k0 < DH; k0 += 32) {
    bf16x16 a0 = load_a_bf16(Q, DH, row0 + lr, k0, hi);
    bf16x16 a1 = load_a_bf16(Q, DH, row0 + 16 + lr, k0, hi);
#pragma unroll
    for (int j = 0; j < 4; ++j) {
      bf16x16 bj = load_b_nt(Kp, DH, col0 + j * 16, k0, lr, hi);
      acc[0][j] = wmma_bf16(a0, bj, acc[0][j]);
      acc[1][j] = wmma_bf16(a1, bj, acc[1][j]);
    }
  }

  const float ninf = -__builtin_inff();
#pragma unroll
  for (int i = 0; i < 2; ++i)
#pragma unroll
    for (int j = 0; j < 4; ++j)
#pragma unroll
      for (int r = 0; r < 8; ++r) {
        int m = row0 + i * 16 + hi * 8 + r;
        int n = col0 + j * 16 + lr;
        float v = fminf(fmaxf(acc[i][j][r], -CLAMP_V), CLAMP_V);
        v = mask[b * S_LEN + n] ? v : ninf;
        attn[((size_t)bh * T_LEN + m) * S_LEN + n] = v;
      }
}

// ---- kernel 4: row softmax over S=2048, in place ---------------------------
__global__ __launch_bounds__(256)
void softmax_rows(float* __restrict__ attn) {
  float* p = attn + (size_t)blockIdx.x * S_LEN;
  const int tid = threadIdx.x;
  const int w = tid >> 5, ln = tid & 31;
  __shared__ float sred[8];

  float vals[8];
  float lmax = -__builtin_inff();
#pragma unroll
  for (int i = 0; i < 8; ++i) { vals[i] = p[tid + i * 256]; lmax = fmaxf(lmax, vals[i]); }
#pragma unroll
  for (int off = 16; off > 0; off >>= 1) lmax = fmaxf(lmax, __shfl_xor(lmax, off, 32));
  if (ln == 0) sred[w] = lmax;
  __syncthreads();
  float m = sred[0];
#pragma unroll
  for (int i = 1; i < 8; ++i) m = fmaxf(m, sred[i]);

  float lsum = 0.f;
#pragma unroll
  for (int i = 0; i < 8; ++i) { vals[i] = __expf(vals[i] - m); lsum += vals[i]; }
#pragma unroll
  for (int off = 16; off > 0; off >>= 1) lsum += __shfl_xor(lsum, off, 32);
  __syncthreads();
  if (ln == 0) sred[w] = lsum;
  __syncthreads();
  float tot = 0.f;
#pragma unroll
  for (int i = 0; i < 8; ++i) tot += sred[i];
  float inv = 1.0f / tot;
#pragma unroll
  for (int i = 0; i < 8; ++i) p[tid + i * 256] = vals[i] * inv;
}

// ---- kernel 5: ctx = P @ V  (P fp32 -> bf16 in-register, V pre-transposed) --
// Vt is [B,NH,DH,S] so this is an NT gemm with contiguous b128 B-loads.
__global__ __launch_bounds__(256)
void attn_av(const float* __restrict__ attn, const __bf16* __restrict__ Vt,
             __bf16* __restrict__ ctx) {
  const int bh = blockIdx.y;
  const int b = bh >> 4, h = bh & 15;
  const int wave = threadIdx.x >> 5, lane = threadIdx.x & 31;
  const int lr = lane & 15, hi = lane >> 4;
  const int row0 = blockIdx.x * 256 + wave * 32;   // 8 waves stacked in M

  const float*  P = attn + (size_t)bh * T_LEN * S_LEN;
  const __bf16* V = Vt   + (size_t)bh * DH * S_LEN;

  f32x8 acc[2][4] = {};
  for (int k0 = 0; k0 < S_LEN; k0 += 32) {
    if (k0 + 32 < S_LEN)
      __builtin_prefetch(P + (size_t)(row0 + lr) * S_LEN + k0 + 32, 0, 3);
    bf16x16 a0 = load_a_f32(P, S_LEN, row0 + lr, k0, hi);
    bf16x16 a1 = load_a_f32(P, S_LEN, row0 + 16 + lr, k0, hi);
#pragma unroll
    for (int j = 0; j < 4; ++j) {
      bf16x16 bj = load_b_nt(V, S_LEN, j * 16, k0, lr, hi);
      acc[0][j] = wmma_bf16(a0, bj, acc[0][j]);
      acc[1][j] = wmma_bf16(a1, bj, acc[1][j]);
    }
  }

#pragma unroll
  for (int i = 0; i < 2; ++i)
#pragma unroll
    for (int j = 0; j < 4; ++j)
#pragma unroll
      for (int r = 0; r < 8; ++r) {
        int t = row0 + i * 16 + hi * 8 + r;
        int d = j * 16 + lr;
        ctx[((size_t)(b * T_LEN + t)) * EMB + h * DH + d] = f2bf(acc[i][j][r]);
      }
}

// ---- host launcher ----------------------------------------------------------
extern "C" void kernel_launch(void* const* d_in, const int* in_sizes, int n_in,
                              void* d_out, int out_size, void* d_ws, size_t ws_size,
                              hipStream_t stream) {
  (void)in_sizes; (void)n_in; (void)out_size; (void)ws_size;

  const float* q    = (const float*)d_in[0];
  const float* k    = (const float*)d_in[1];
  const float* v    = (const float*)d_in[2];
  const int*   mask = (const int*)  d_in[3];
  const float* Wq   = (const float*)d_in[4];
  const float* bq   = (const float*)d_in[5];
  const float* Wk   = (const float*)d_in[6];
  const float* bk   = (const float*)d_in[7];
  const float* Wv   = (const float*)d_in[8];
  const float* bv   = (const float*)d_in[9];
  const float* Wo   = (const float*)d_in[10];
  const float* bo   = (const float*)d_in[11];

  float* outF = (float*)d_out;                                  // [B,T,EMB]
  float* attn = outF + (size_t)B_SZ * T_LEN * EMB;              // [B,NH,T,S]

  char* wp = (char*)d_ws;
  size_t off = 0;
  auto alloc = [&](size_t bytes) -> void* {
    void* p = wp + off;
    off += (bytes + 255) & ~(size_t)255;
    return p;
  };
  const size_t BT = (size_t)B_SZ * T_LEN;                       // 4096
  __bf16* qb  = (__bf16*)alloc(BT * EMB * 2);
  __bf16* kb  = (__bf16*)alloc(BT * EMB * 2);
  __bf16* vb  = (__bf16*)alloc(BT * EMB * 2);
  __bf16* Wqb = (__bf16*)alloc((size_t)EMB * EMB * 2);
  __bf16* Wkb = (__bf16*)alloc((size_t)EMB * EMB * 2);
  __bf16* Wvb = (__bf16*)alloc((size_t)EMB * EMB * 2);
  __bf16* Wob = (__bf16*)alloc((size_t)EMB * EMB * 2);
  __bf16* Qh  = (__bf16*)alloc(BT * EMB * 2);                   // [B,NH,T,DH] (pre-scaled)
  __bf16* Kh  = (__bf16*)alloc(BT * EMB * 2);                   // [B,NH,S,DH]
  __bf16* Vt  = (__bf16*)alloc(BT * EMB * 2);                   // [B,NH,DH,S]
  __bf16* ctx = (__bf16*)alloc(BT * EMB * 2);                   // [B,T,EMB]

  const int nX = (int)(BT * EMB);                               // 4,194,304
  const int nW = EMB * EMB;                                     // 1,048,576
  cvt_f32_bf16<<<nX / 256, 256, 0, stream>>>(q,  qb,  nX);
  cvt_f32_bf16<<<nX / 256, 256, 0, stream>>>(k,  kb,  nX);
  cvt_f32_bf16<<<nX / 256, 256, 0, stream>>>(v,  vb,  nX);
  cvt_f32_bf16<<<nW / 256, 256, 0, stream>>>(Wq, Wqb, nW);
  cvt_f32_bf16<<<nW / 256, 256, 0, stream>>>(Wk, Wkb, nW);
  cvt_f32_bf16<<<nW / 256, 256, 0, stream>>>(Wv, Wvb, nW);
  cvt_f32_bf16<<<nW / 256, 256, 0, stream>>>(Wo, Wob, nW);

  dim3 gProj((unsigned)((BT / 128) * (EMB / 128)));             // 256 blocks
  gemm_bf16_nt<<<gProj, 256, 0, stream>>>(qb, Wqb, bq, SCALE_F, (int)BT, EMB, EMB, nullptr, Qh, 1);
  gemm_bf16_nt<<<gProj, 256, 0, stream>>>(kb, Wkb, bk, 1.0f,    (int)BT, EMB, EMB, nullptr, Kh, 1);
  gemm_bf16_nt<<<gProj, 256, 0, stream>>>(vb, Wvb, bv, 1.0f,    (int)BT, EMB, EMB, nullptr, Vt, 2);

  dim3 gScore((T_LEN / 128) * (S_LEN / 128), B_SZ * NH);        // (256, 32)
  attn_scores<<<gScore, 256, 0, stream>>>(Qh, Kh, mask, attn);

  softmax_rows<<<dim3(B_SZ * NH * T_LEN), 256, 0, stream>>>(attn);

  attn_av<<<dim3(T_LEN / 256, B_SZ * NH), 256, 0, stream>>>(attn, Vt, ctx);

  gemm_bf16_nt<<<gProj, 256, 0, stream>>>(ctx, Wob, bo, 1.0f, (int)BT, EMB, EMB, outF, nullptr, 0);
}
